// BayesianSG_49916109914289
// MI455X (gfx1250) — matrix operand order repaired and checked by reading, not compiled, not run on hardware
//
#include <hip/hip_runtime.h>
#include <hip/hip_bf16.h>

typedef __attribute__((ext_vector_type(16))) _Float16 v16h;
typedef __attribute__((ext_vector_type(8)))  float    v8f;
typedef int v4i_vs __attribute__((vector_size(16)));   // matches builtin param type

#define VOCAB 50000
#define DIM   256
#define BATCH 256
#define CTX   10
#define NEG_BIG (-1.0e30f)

// ---------------------------------------------------------------------------
// WMMA fragment loaders: fp32 memory -> f16 fragments (cvt in registers).
// Layouts per CDNA5 ISA 7.12.2 (wave32):
//   A 16x32 f16 : lane L holds row m=L&15; half j (j<8)  = A[m][kb + 8*hi + j]
//                 half 8+j = A[m][kb + 16 + 8*hi + j]          (hi = L>>4)
//   B 32x16 f16 : lane L holds col n=L&15; half j (j<8) = W[n][kb+8*hi+j], half 8+j = W[n][kb+16+8*hi+j]
//   C 16x16 f32 : lane L col n=L&15; VGPR i holds row m = i + 8*hi
// ---------------------------------------------------------------------------
__device__ __forceinline__ v16h load_frag_a(const float* __restrict__ row, int kb, int hi) {
  v16h a;
  const float* p0 = row + kb + 8 * hi;
  const float* p1 = row + kb + 16 + 8 * hi;
#pragma unroll
  for (int j = 0; j < 8; ++j) {
    a[j]     = (_Float16)p0[j];
    a[8 + j] = (_Float16)p1[j];
  }
  return a;
}

__device__ __forceinline__ v16h load_frag_b(const float* __restrict__ wrow, int kb, int hi) {
  v16h b;
  const float* p0 = wrow + kb + 8 * hi;
  const float* p1 = wrow + kb + 16 + 8 * hi;
#pragma unroll
  for (int j = 0; j < 8; ++j) { b[j] = (_Float16)p0[j]; b[8 + j] = (_Float16)p1[j]; }
  return b;
}

__device__ __forceinline__ float softplus_f(float x) {
  return fmaxf(x, 0.0f) + log1pf(expf(-fabsf(x)));
}

// Async global->LDS staging (CDNA5 ASYNCcnt path) with safe fallback.
__device__ __forceinline__ void stage16(const float* src, float* dst) {
#if __has_builtin(__builtin_amdgcn_global_load_async_to_lds_b128)
  __builtin_amdgcn_global_load_async_to_lds_b128(
      (__attribute__((address_space(1))) v4i_vs*)src,
      (__attribute__((address_space(3))) v4i_vs*)dst, 0, 0);
#else
  *(float4*)dst = *(const float4*)src;
#endif
}

__device__ __forceinline__ void stage_wait() {
#if __has_builtin(__builtin_amdgcn_global_load_async_to_lds_b128)
#if __has_builtin(__builtin_amdgcn_s_wait_asynccnt)
  __builtin_amdgcn_s_wait_asynccnt(0);
#else
  asm volatile("s_wait_asynccnt 0" ::: "memory");
#endif
#endif
}

// ---------------------------------------------------------------------------
// Kernel 1: encoder GEMM.  hfull[m][e] = relu(enc_in[m] . enc_W[e] + enc_b[e])
// m = b*10+c ; enc_in row = [emb[center[b]] | emb[ctx[b][c]]]  (gathered on load)
// grid = 160 Mtiles * 4 Nchunks ; block = 256 (8 waves, each 16 N columns)
// ---------------------------------------------------------------------------
__global__ __launch_bounds__(256) void enc_kernel(
    const int* __restrict__ center_id, const int* __restrict__ context_ids,
    const float* __restrict__ emb, const float* __restrict__ encW,
    const float* __restrict__ encb, float* __restrict__ hfull) {
  const int tid = threadIdx.x, wave = tid >> 5, lane = tid & 31;
  const int l15 = lane & 15, hi = lane >> 4;
  const int mt = blockIdx.x >> 2, nc = blockIdx.x & 3;
  const int n = nc * 128 + wave * 16 + l15;          // output column e (<512)
  const int m = mt * 16 + l15;                       // A row this lane supplies
  const int b = m / CTX, c = m % CTX;
  const float* rowC = emb + (size_t)center_id[b] * DIM;
  const float* rowX = emb + (size_t)context_ids[b * CTX + c] * DIM;
  const float* wrow = encW + (size_t)n * 512;

  v8f acc = {};
#pragma unroll
  for (int kb8 = 0; kb8 < 16; ++kb8) {
    const int kb = kb8 * 32;
    const int k0 = kb + 8 * hi;
    const int k1 = kb + 16 + 8 * hi;
    const float* p0 = (k0 < DIM) ? (rowC + k0) : (rowX + (k0 - DIM));
    const float* p1 = (k1 < DIM) ? (rowC + k1) : (rowX + (k1 - DIM));
    v16h a;
#pragma unroll
    for (int j = 0; j < 8; ++j) { a[j] = (_Float16)p0[j]; a[8 + j] = (_Float16)p1[j]; }
    v16h bf = load_frag_b(wrow, kb, hi);
    acc = __builtin_amdgcn_wmma_f32_16x16x32_f16(false, a, false, bf, (short)0, acc,
                                                 false, false);
  }
  const float bias = encb[n];
#pragma unroll
  for (int i = 0; i < 8; ++i) {
    const int mg = mt * 16 + i + 8 * hi;
    float v = acc[i] + bias;
    hfull[(size_t)mg * 512 + n] = v > 0.0f ? v : 0.0f;
  }
}

// Kernel 2: h_sum[b][e] = sum_c hfull[b*10+c][e]   (deterministic fixed-order)
__global__ __launch_bounds__(256) void hred_kernel(const float* __restrict__ hfull,
                                                   float* __restrict__ h_sum) {
  const int idx = blockIdx.x * 256 + threadIdx.x;   // < 256*512
  const int b = idx >> 9, e = idx & 511;
  float s = 0.0f;
#pragma unroll
  for (int c = 0; c < CTX; ++c) s += hfull[(size_t)(b * CTX + c) * 512 + e];
  h_sum[idx] = s;
}

// ---------------------------------------------------------------------------
// Kernel 3: mean|var GEMM.  mv[b][n] = h_sum[b] . W2[n] + bias2[n]
// n<256 -> mean_W/mean_b ; n>=256 -> var_W/var_b.  grid = 16 Mtiles * 4 Nchunks
// ---------------------------------------------------------------------------
__global__ __launch_bounds__(256) void mv_kernel(
    const float* __restrict__ h_sum, const float* __restrict__ meanW,
    const float* __restrict__ meanb, const float* __restrict__ varW,
    const float* __restrict__ varb, float* __restrict__ mv) {
  const int tid = threadIdx.x, wave = tid >> 5, lane = tid & 31;
  const int l15 = lane & 15, hi = lane >> 4;
  const int mt = blockIdx.x >> 2, nc = blockIdx.x & 3;
  const int n = nc * 128 + wave * 16 + l15;
  const float* wrow = (n < DIM) ? (meanW + (size_t)n * 512) : (varW + (size_t)(n - DIM) * 512);
  const float  bias = (n < DIM) ? meanb[n] : varb[n - DIM];
  const float* arow = h_sum + (size_t)(mt * 16 + l15) * 512;

  v8f acc = {};
#pragma unroll
  for (int kb8 = 0; kb8 < 16; ++kb8) {
    v16h a  = load_frag_a(arow, kb8 * 32, hi);
    v16h bf = load_frag_b(wrow, kb8 * 32, hi);
    acc = __builtin_amdgcn_wmma_f32_16x16x32_f16(false, a, false, bf, (short)0, acc,
                                                 false, false);
  }
#pragma unroll
  for (int i = 0; i < 8; ++i)
    mv[(size_t)(mt * 16 + i + 8 * hi) * 512 + n] = acc[i] + bias;
}

// Kernel 4: z = mean + exp(var/2)*eps ; per-row KL.  one block per b.
__global__ __launch_bounds__(256) void zkl_kernel(
    const int* __restrict__ center_id, const float* __restrict__ prior_means,
    const float* __restrict__ prior_vars, const float* __restrict__ mv,
    const float* __restrict__ eps, float* __restrict__ z, float* __restrict__ kl) {
  __shared__ float red[256];
  const int b = blockIdx.x, d = threadIdx.x;
  const float mean = mv[(size_t)b * 512 + d];
  const float var  = softplus_f(mv[(size_t)b * 512 + DIM + d]);
  z[(size_t)b * DIM + d] = mean + expf(var * 0.5f) * eps[d];
  const int cid = center_id[b];
  const float pm = prior_means[(size_t)cid * DIM + d];
  const float pv = softplus_f(prior_vars[(size_t)cid * DIM + d]);
  const float dm = pm - mean;
  red[d] = var / pv + dm * dm / pv + logf(pv) - logf(var);
  __syncthreads();
  for (int off = 128; off > 0; off >>= 1) {
    if (d < off) red[d] += red[d + off];
    __syncthreads();
  }
  if (d == 0) kl[b] = 0.5f * (red[0] - (float)DIM);
}

// ---------------------------------------------------------------------------
// Kernel 5: logits GEMM + fused online-softmax partials.
// grid = 391 blocks of 128 columns; each wave holds B fragments for full K=256
// (vocab_W read from HBM exactly once) and loops all 16 M tiles.
// A tile (16x256 f32) is staged once per block into LDS via async global->LDS
// (ASYNCcnt), killing the 8x redundant per-wave A reads.  Per-row softmax stats
// are reduced with wave32 shuffle butterflies, then 8 wave-partials merge in LDS.
// ---------------------------------------------------------------------------
__global__ __launch_bounds__(256) void logits_kernel(
    const float* __restrict__ z, const float* __restrict__ vocabW,
    const float* __restrict__ vocabb, float* __restrict__ rmax,
    float* __restrict__ rsum) {
  __shared__ float alds[16][260];                    // padded: stride%64 = 4 banks
  __shared__ float wmaxs[8][16];
  __shared__ float wsums[8][16];
  const int tid = threadIdx.x, wave = tid >> 5, lane = tid & 31;
  const int l15 = lane & 15, hi = lane >> 4;
  const int n = blockIdx.x * 128 + wave * 16 + l15;  // global vocab column
  const int nclamp = n < VOCAB ? n : (VOCAB - 1);
  const bool valid = n < VOCAB;
  const float* wrow = vocabW + (size_t)nclamp * DIM;
  const float  bias = vocabb[nclamp];

  v16h Bf[8];
#pragma unroll
  for (int kb8 = 0; kb8 < 8; ++kb8) Bf[kb8] = load_frag_b(wrow, kb8 * 32, hi);

  for (int mt = 0; mt < 16; ++mt) {
    // --- stage A tile (16 rows x 256 f32) into LDS, 16B per thread x 4 ---
#pragma unroll
    for (int it = 0; it < 4; ++it) {
      const int fo  = (it * 256 + tid) * 4;          // flat float offset
      const int row = fo >> 8, col = fo & 255;
      stage16(z + (size_t)(mt * 16 + row) * DIM + col, &alds[row][col]);
    }
    stage_wait();
    __syncthreads();                                 // B1: tile visible

    v8f acc = {};
#pragma unroll
    for (int kb8 = 0; kb8 < 8; ++kb8) {
      const int kb = kb8 * 32;
      v16h a;
      const float* p0 = &alds[l15][kb + 8 * hi];
      const float* p1 = &alds[l15][kb + 16 + 8 * hi];
#pragma unroll
      for (int j = 0; j < 8; ++j) { a[j] = (_Float16)p0[j]; a[8 + j] = (_Float16)p1[j]; }
      acc = __builtin_amdgcn_wmma_f32_16x16x32_f16(false, a, false, Bf[kb8], (short)0,
                                                   acc, false, false);
    }

    // --- per-row (max, sum exp) over this wave's 16 columns via shuffles ---
#pragma unroll
    for (int i = 0; i < 8; ++i) {
      float v = valid ? (acc[i] + bias) : NEG_BIG;
      float mx = v;
#pragma unroll
      for (int off = 1; off < 16; off <<= 1) mx = fmaxf(mx, __shfl_xor(mx, off, 32));
      float e = expf(v - mx);
      float sm = e;
#pragma unroll
      for (int off = 1; off < 16; off <<= 1) sm += __shfl_xor(sm, off, 32);
      if (l15 == i) {                                // one writer per (wave,row)
        wmaxs[wave][i + 8 * hi] = mx;
        wsums[wave][i + 8 * hi] = sm;
      }
    }
    __syncthreads();                                 // B2: wave partials visible

    if (tid < 16) {                                  // merge 8 wave partials/row
      float m_ = wmaxs[0][tid];
#pragma unroll
      for (int w = 1; w < 8; ++w) m_ = fmaxf(m_, wmaxs[w][tid]);
      float s = 0.0f;
#pragma unroll
      for (int w = 0; w < 8; ++w) s += wsums[w][tid] * expf(wmaxs[w][tid] - m_);
      const int rg = mt * 16 + tid;
      rmax[(size_t)blockIdx.x * BATCH + rg] = m_;
      rsum[(size_t)blockIdx.x * BATCH + rg] = s;
    }
    __syncthreads();                                 // B3: LDS reusable
  }
}

// Kernel 6: gathered logits for context ids. one wave per (b,c); 320 blocks.
__global__ __launch_bounds__(256) void ctx_kernel(
    const int* __restrict__ context_ids, const float* __restrict__ z,
    const float* __restrict__ vocabW, const float* __restrict__ vocabb,
    float* __restrict__ logit_ctx) {
  const int tid = threadIdx.x, wave = tid >> 5, lane = tid & 31;
  const int task = blockIdx.x * 8 + wave;            // < 2560
  const int b = task / CTX;
  const int id = context_ids[task];
  const float* zr = z + (size_t)b * DIM;
  const float* wr = vocabW + (size_t)id * DIM;
  float s = 0.0f;
#pragma unroll
  for (int j = 0; j < 8; ++j) s += zr[lane + 32 * j] * wr[lane + 32 * j];
#pragma unroll
  for (int off = 16; off > 0; off >>= 1) s += __shfl_xor(s, off, 32);
  if (lane == 0) logit_ctx[task] = s + vocabb[id];
}

// Kernel 7: merge softmax partials -> logZ[b]. 1 block, thread = b.
__global__ __launch_bounds__(256) void logz_kernel(const float* __restrict__ rmax,
                                                   const float* __restrict__ rsum,
                                                   int nblocks, float* __restrict__ logZ) {
  const int b = threadIdx.x;
  float m_ = -__builtin_inff(), s = 0.0f;
  for (int i = 0; i < nblocks; ++i) {
    const float pm = rmax[(size_t)i * BATCH + b];
    const float ps = rsum[(size_t)i * BATCH + b];
    if (pm > m_) { s = s * expf(m_ - pm) + ps; m_ = pm; }
    else         { s += ps * expf(pm - m_); }
  }
  logZ[b] = m_ + logf(s);
}

// Kernel 8: out = sum_b ( sum_c logit_ctx - C*logZ[b] - kl[b] )
__global__ __launch_bounds__(256) void final_kernel(const float* __restrict__ logit_ctx,
                                                    const float* __restrict__ logZ,
                                                    const float* __restrict__ kl,
                                                    float* __restrict__ out) {
  __shared__ float red[256];
  const int b = threadIdx.x;
  float r = 0.0f;
#pragma unroll
  for (int c = 0; c < CTX; ++c) r += logit_ctx[b * CTX + c];
  red[b] = r - (float)CTX * logZ[b] - kl[b];
  __syncthreads();
  for (int off = 128; off > 0; off >>= 1) {
    if (b < off) red[b] += red[b + off];
    __syncthreads();
  }
  if (b == 0) out[0] = red[0];
}

// ---------------------------------------------------------------------------
extern "C" void kernel_launch(void* const* d_in, const int* in_sizes, int n_in,
                              void* d_out, int out_size, void* d_ws, size_t ws_size,
                              hipStream_t stream) {
  (void)in_sizes; (void)n_in; (void)out_size; (void)ws_size;
  const int*   center_id   = (const int*)d_in[0];
  const int*   context_ids = (const int*)d_in[1];
  const float* embeddings  = (const float*)d_in[2];
  const float* prior_means = (const float*)d_in[3];
  const float* prior_vars  = (const float*)d_in[4];
  const float* enc_W       = (const float*)d_in[5];
  const float* enc_b       = (const float*)d_in[6];
  const float* mean_W      = (const float*)d_in[7];
  const float* mean_b      = (const float*)d_in[8];
  const float* var_W       = (const float*)d_in[9];
  const float* var_b       = (const float*)d_in[10];
  const float* vocab_W     = (const float*)d_in[11];
  const float* vocab_b     = (const float*)d_in[12];
  const float* epsilon     = (const float*)d_in[13];
  float* out = (float*)d_out;

  const int NB = (VOCAB + 127) / 128;                // 391
  float* ws = (float*)d_ws;
  float* hfull     = ws;                  // 2560*512
  float* h_sum     = hfull + 2560 * 512;  // 256*512
  float* mv        = h_sum + 256 * 512;   // 256*512
  float* zbuf      = mv + 256 * 512;      // 256*256
  float* kl        = zbuf + 256 * 256;    // 256
  float* logit_ctx = kl + 256;            // 2560
  float* logZ      = logit_ctx + 2560;    // 256
  float* rmax      = logZ + 256;          // NB*256
  float* rsum      = rmax + NB * 256;     // NB*256

  enc_kernel<<<160 * 4, 256, 0, stream>>>(center_id, context_ids, embeddings, enc_W,
                                          enc_b, hfull);
  hred_kernel<<<(256 * 512) / 256, 256, 0, stream>>>(hfull, h_sum);
  mv_kernel<<<16 * 4, 256, 0, stream>>>(h_sum, mean_W, mean_b, var_W, var_b, mv);
  zkl_kernel<<<BATCH, 256, 0, stream>>>(center_id, prior_means, prior_vars, mv, epsilon,
                                        zbuf, kl);
  logits_kernel<<<NB, 256, 0, stream>>>(zbuf, vocab_W, vocab_b, rmax, rsum);
  ctx_kernel<<<2560 / 8, 256, 0, stream>>>(context_ids, zbuf, vocab_W, vocab_b,
                                           logit_ctx);
  logz_kernel<<<1, 256, 0, stream>>>(rmax, rsum, NB, logZ);
  final_kernel<<<1, 256, 0, stream>>>(logit_ctx, logZ, kl, out);
}
